// HierarchicalHybridMatformer_40681930228131
// MI455X (gfx1250) — compile-verified, matrix operand order checked by reading, not run on hardware
//
#include <hip/hip_runtime.h>
#include <stdint.h>

// ---------------------------------------------------------------------------
// HierarchicalHybridMatformer forward on MI455X (gfx1250, wave32, WMMA).
// All dense layers go through a bf16 WMMA GEMM (f32 accumulate); graph ops
// (gather / segment softmax / scatter-add / pooling) are elementwise kernels.
// ---------------------------------------------------------------------------

#define NN 8000      // nodes
#define NE 32000     // edges
#define NG 64        // graphs
#define CH 256       // hidden
#define NH 4         // heads
#define HC 1024      // NH*CH
#define TC 768       // 3*CH
#define EDIM 128

typedef __attribute__((ext_vector_type(16))) __bf16        v16bf;
typedef __attribute__((ext_vector_type(8)))  float         v8f;
typedef __attribute__((ext_vector_type(4)))  unsigned int  v4u;
typedef __attribute__((ext_vector_type(8)))  unsigned int  v8u;

__device__ __forceinline__ unsigned short f2bf(float f) {
  unsigned u = __float_as_uint(f);
  unsigned r = u + 0x7fffu + ((u >> 16) & 1u);   // round-to-nearest-even
  return (unsigned short)(r >> 16);
}
// order-preserving float<->uint key for atomic max on floats
__device__ __forceinline__ unsigned fkey(float f) {
  unsigned u = __float_as_uint(f);
  return (u & 0x80000000u) ? ~u : (u | 0x80000000u);
}
__device__ __forceinline__ float funkey(unsigned k) {
  unsigned u = (k & 0x80000000u) ? (k & 0x7fffffffu) : ~k;
  return __uint_as_float(u);
}
__device__ __forceinline__ float leakyf(float x) { return x > 0.f ? x : 0.01f * x; }

// ---------------------------------------------------------------------------
// Generic GEMM: C[M,N](ldc) = act( A[M,K] @ B[K,N] + bias )
// bf16 WMMA 16x16x32, BM=128 BN=64 BK=32, 8 waves (256 thr) per block,
// each wave computes a 32x32 patch = 2x2 WMMA tiles.
//
// LDS layout: each lane's WMMA fragment (ISA map k = (e>>3)*16 + half*8 +
// (e&7)) is exactly two contiguous 16-byte runs:
//   A: sA[row][k]  (row stride 40 ushorts = 80 B, 16-B aligned runs)
//   B: stored transposed, sBt[col][k], same stride
// -> fragment build = 2x ds_load_b128 per fragment (no scratch).
// Staging: fast interior path = float4 global loads + packed bf16 stores;
// guarded scalar path only for M/K/N tail tiles.
// ---------------------------------------------------------------------------
#define BM 128
#define BN 64
#define BK 32
#define LSTR 40   // LDS row stride in ushorts (80 B: 16-B aligned, conflict-free)

__device__ __forceinline__ v16bf frag_ld(const unsigned short* rowp, int half) {
  const v4u lo = *(const v4u*)(rowp + half * 8);
  const v4u hi = *(const v4u*)(rowp + 16 + half * 8);
  v8u w;
  w[0] = lo[0]; w[1] = lo[1]; w[2] = lo[2]; w[3] = lo[3];
  w[4] = hi[0]; w[5] = hi[1]; w[6] = hi[2]; w[7] = hi[3];
  return __builtin_bit_cast(v16bf, w);
}

__global__ __launch_bounds__(256) void k_gemm(
    const float* __restrict__ A, const float* __restrict__ B,
    const float* __restrict__ bias, float* __restrict__ C,
    int M, int N, int K, int ldc, int act)
{
  __shared__ unsigned short sA[BM][LSTR];
  __shared__ unsigned short sBt[BN][LSTR];

  const int tid  = threadIdx.x;
  const int lane = tid & 31;
  const int wave = tid >> 5;
  const int half = lane >> 4;   // 0/1: which 16-lane group
  const int idx  = lane & 15;
  const int wrow = wave >> 1;   // 0..3 : 32-row band
  const int wcol = wave & 1;    // 0..1 : 32-col band

  const int rowBase = blockIdx.y * BM;
  const int colBase = blockIdx.x * BN;

  const bool aInterior = (rowBase + BM <= M) && ((K & 3) == 0);
  const bool bInterior = (colBase + BN <= N) && ((N & 3) == 0);

  v8f acc[2][2];
  for (int r = 0; r < 2; ++r)
    for (int c = 0; c < 2; ++c)
      for (int i = 0; i < 8; ++i) acc[r][c][i] = 0.f;

  for (int k0 = 0; k0 < K; k0 += BK) {
    const bool kFull = (k0 + BK <= K);

    // ---- stage A tile (fp32 -> bf16) ----
    if (aInterior && kFull) {
      // 1024 float4s / 256 threads: b128 loads, packed b64 LDS stores
      #pragma unroll
      for (int ii = 0; ii < 4; ++ii) {
        int f = tid + ii * 256;          // float4 index in tile
        int r = f >> 3, qd = f & 7;      // 8 float4 per 32-wide row
        const float4 av = *(const float4*)(A + (size_t)(rowBase + r) * K + k0 + qd * 4);
        uint2 pk;
        pk.x = (unsigned)f2bf(av.x) | ((unsigned)f2bf(av.y) << 16);
        pk.y = (unsigned)f2bf(av.z) | ((unsigned)f2bf(av.w) << 16);
        *(uint2*)(&sA[r][qd * 4]) = pk;  // 8-B aligned (80*r + 8*qd)
      }
    } else {
      for (int i = tid; i < BM * BK; i += 256) {
        int r = i >> 5, kk = i & 31;
        int gr = rowBase + r, gk = k0 + kk;
        float fv = (gr < M && gk < K) ? A[(size_t)gr * K + gk] : 0.f;
        sA[r][kk] = f2bf(fv);
      }
    }

    // ---- stage B tile transposed: sBt[col][k] ----
    if (bInterior && kFull) {
      #pragma unroll
      for (int ii = 0; ii < 2; ++ii) {
        int f = tid + ii * 256;          // 512 float4s
        int kk = f >> 4, qd = f & 15;    // 16 float4 per 64-wide row
        const float4 bv = *(const float4*)(B + (size_t)(k0 + kk) * N + colBase + qd * 4);
        sBt[qd * 4 + 0][kk] = f2bf(bv.x);
        sBt[qd * 4 + 1][kk] = f2bf(bv.y);
        sBt[qd * 4 + 2][kk] = f2bf(bv.z);
        sBt[qd * 4 + 3][kk] = f2bf(bv.w);
      }
    } else {
      for (int i = tid; i < BK * BN; i += 256) {
        int kk = i >> 6, c = i & 63;
        int gk = k0 + kk, gc = colBase + c;
        float fv = (gk < K && gc < N) ? B[(size_t)gk * N + gc] : 0.f;
        sBt[c][kk] = f2bf(fv);
      }
    }

    // prefetch next A tile into cache while we compute
    if (k0 + BK < K) {
      int pr = rowBase + (tid >> 1);
      if (pr < M) __builtin_prefetch(&A[(size_t)pr * K + (k0 + BK)], 0, 1);
    }
    __syncthreads();

    v16bf af0 = frag_ld(&sA[wrow * 32 + idx][0], half);
    v16bf af1 = frag_ld(&sA[wrow * 32 + 16 + idx][0], half);
    v16bf bf0 = frag_ld(&sBt[wcol * 32 + idx][0], half);
    v16bf bf1 = frag_ld(&sBt[wcol * 32 + 16 + idx][0], half);

    acc[0][0] = __builtin_amdgcn_wmma_f32_16x16x32_bf16(
        false, af0, false, bf0, (short)0, acc[0][0], false, false);
    acc[0][1] = __builtin_amdgcn_wmma_f32_16x16x32_bf16(
        false, af0, false, bf1, (short)0, acc[0][1], false, false);
    acc[1][0] = __builtin_amdgcn_wmma_f32_16x16x32_bf16(
        false, af1, false, bf0, (short)0, acc[1][0], false, false);
    acc[1][1] = __builtin_amdgcn_wmma_f32_16x16x32_bf16(
        false, af1, false, bf1, (short)0, acc[1][1], false, false);
    __syncthreads();
  }

  // epilogue: C/D layout -> VGPR i holds row (i + 8*half), col = idx
  for (int r = 0; r < 2; ++r)
    for (int c = 0; c < 2; ++c) {
      int col = colBase + wcol * 32 + c * 16 + idx;
      if (col >= N) continue;
      float bv = bias ? bias[col] : 0.f;
      #pragma unroll
      for (int i = 0; i < 8; ++i) {
        int row = rowBase + wrow * 32 + r * 16 + half * 8 + i;
        if (row >= M) continue;
        float v = acc[r][c][i] + bv;
        if (act == 1) v = leakyf(v);
        C[(size_t)row * ldc + col] = v;
      }
    }
}

// ---------------------------------------------------------------------------
// Elementwise / graph kernels
// ---------------------------------------------------------------------------
__global__ void k_zero(float* p, size_t n) {
  size_t t = (size_t)blockIdx.x * blockDim.x + threadIdx.x;
  if (t < n) p[t] = 0.f;
}

__global__ void k_embed(const int* __restrict__ x, const float* __restrict__ emb,
                        float* __restrict__ h, int total) {
  int t = blockIdx.x * blockDim.x + threadIdx.x;
  if (t >= total) return;
  h[t] = emb[(size_t)x[t >> 8] * CH + (t & 255)];
}

__global__ void k_rbf(const float* __restrict__ ea, float* __restrict__ rbf, int total) {
  int t = blockIdx.x * blockDim.x + threadIdx.x;
  if (t >= total) return;
  int e = t >> 7, j = t & 127;
  float d = ea[e];
  float ctr = 8.f * (float)j / 127.f;   // linspace(0,8,128)
  float sg = 8.f / 128.f;
  float z = (d - ctr) / sg;
  rbf[t] = expf(-z * z);
}

// triplet attention logits: one wave per (edge, head)
__global__ __launch_bounds__(256) void k_alpha(
    const float* __restrict__ q, const float* __restrict__ km,
    const float* __restrict__ ef, const int* __restrict__ src,
    const int* __restrict__ dst, float* __restrict__ araw,
    unsigned* __restrict__ amax, int EH)
{
  int w = (int)((blockIdx.x * blockDim.x + threadIdx.x) >> 5);
  int lane = threadIdx.x & 31;
  if (w >= EH) return;
  int e = w >> 2, h = w & 3;
  int s = src[e], d = dst[e];
  const float* qp = q + (size_t)d * HC + h * CH;
  const float* kd = km + (size_t)d * HC + h * CH;
  const float* ks = km + (size_t)s * HC + h * CH;
  const float* ep = ef + (size_t)e * HC + h * CH;
  float sum = 0.f;
  for (int c = lane; c < CH; c += 32) sum += qp[c] * (kd[c] + ks[c] + ep[c]);
  for (int off = 16; off; off >>= 1) sum += __shfl_down(sum, off, 32);
  if (lane == 0) {
    float a = sum * 0.03608439182435161f;  // 1/sqrt(3*C)
    araw[w] = a;
    atomicMax(amax + (size_t)d * NH + h, fkey(a));
  }
}

__global__ void k_expsum(float* __restrict__ a, const int* __restrict__ dst,
                         const unsigned* __restrict__ amax, float* __restrict__ asum, int EH) {
  int t = blockIdx.x * blockDim.x + threadIdx.x;
  if (t >= EH) return;
  int e = t >> 2, h = t & 3;
  int d = dst[e];
  float v = expf(a[t] - funkey(amax[(size_t)d * NH + h]));
  a[t] = v;
  atomicAdd(asum + (size_t)d * NH + h, v);
}

__global__ void k_anorm(float* __restrict__ a, const int* __restrict__ dst,
                        const float* __restrict__ asum, int EH) {
  int t = blockIdx.x * blockDim.x + threadIdx.x;
  if (t >= EH) return;
  int e = t >> 2, h = t & 3;
  a[t] = a[t] / (asum[(size_t)dst[e] * NH + h] + 1e-16f);
}

// m[e,h,:] = concat(v[dst], v[src], e_proj) * alpha
__global__ void k_build_msg(const float* __restrict__ vv, const float* __restrict__ ef,
                            const float* __restrict__ alpha, const int* __restrict__ src,
                            const int* __restrict__ dst, float* __restrict__ m, int total) {
  int t = blockIdx.x * blockDim.x + threadIdx.x;
  if (t >= total) return;
  int e = t >> 10, r = t & 1023, h = r >> 8, c = r & 255;
  float al = alpha[(size_t)e * NH + h];
  int s = src[e], d = dst[e];
  size_t base = ((size_t)e * NH + h) * TC;
  m[base + c]            = vv[(size_t)d * HC + h * CH + c] * al;
  m[base + CH + c]       = vv[(size_t)s * HC + h * CH + c] * al;
  m[base + 2 * CH + c]   = ef[(size_t)e * HC + h * CH + c] * al;
}

__global__ void k_scatter(const float* __restrict__ m, const int* __restrict__ dst,
                          float* __restrict__ agg, int total) {
  int t = blockIdx.x * blockDim.x + threadIdx.x;
  if (t >= total) return;
  int e = t >> 10, r = t & 1023, h = r >> 8, c = r & 255;
  atomicAdd(&agg[(size_t)dst[e] * HC + h * CH + c], m[((size_t)e * NH + h) * CH + c]);
}

// row LayerNorm (C_ <= 256), optional fused LeakyReLU, distinct in/out strides
__global__ __launch_bounds__(256) void k_ln(
    const float* __restrict__ x, float* __restrict__ y,
    const float* __restrict__ g, const float* __restrict__ b,
    int C_, int ldx, int ldy, int leaky)
{
  __shared__ float red[256];
  int row = blockIdx.x, tid = threadIdx.x;
  const float* xr = x + (size_t)row * ldx;
  float v = (tid < C_) ? xr[tid] : 0.f;
  red[tid] = v; __syncthreads();
  for (int s = 128; s; s >>= 1) { if (tid < s) red[tid] += red[tid + s]; __syncthreads(); }
  float mean = red[0] / (float)C_;
  __syncthreads();
  float dv = (tid < C_) ? (v - mean) : 0.f;
  red[tid] = dv * dv; __syncthreads();
  for (int s = 128; s; s >>= 1) { if (tid < s) red[tid] += red[tid + s]; __syncthreads(); }
  float var = red[0] / (float)C_;
  if (tid < C_) {
    float o = dv * rsqrtf(var + 1e-5f) * g[tid] + b[tid];
    if (leaky) o = leakyf(o);
    y[(size_t)row * ldy + tid] = o;
  }
}

// h = silu(batchnorm(o)) + skip
__global__ void k_bnsilu(const float* __restrict__ o, const float* __restrict__ skip,
                         float* __restrict__ h, const float* __restrict__ g,
                         const float* __restrict__ b, const float* __restrict__ m,
                         const float* __restrict__ v, int total) {
  int t = blockIdx.x * blockDim.x + threadIdx.x;
  if (t >= total) return;
  int c = t & 255;
  float z = (o[t] - m[c]) * rsqrtf(v[c] + 1e-5f) * g[c] + b[c];
  z = z / (1.f + expf(-z));
  h[t] = z + skip[t];
}

__global__ void k_vxinit(const float* __restrict__ vemb, float* __restrict__ vx, int total) {
  int t = blockIdx.x * blockDim.x + threadIdx.x;
  if (t < total) vx[t] = vemb[t & 255];
}

__global__ void k_vxacc(const float* __restrict__ h, const int* __restrict__ batch,
                        float* __restrict__ vx, int total) {
  int t = blockIdx.x * blockDim.x + threadIdx.x;
  if (t >= total) return;
  atomicAdd(&vx[(size_t)batch[t >> 8] * CH + (t & 255)], h[t]);
}

__global__ void k_addvx(const float* __restrict__ h, const float* __restrict__ vx,
                        const int* __restrict__ batch, float* __restrict__ y, int total) {
  int t = blockIdx.x * blockDim.x + threadIdx.x;
  if (t >= total) return;
  y[t] = h[t] + vx[(size_t)batch[t >> 8] * CH + (t & 255)];
}

__global__ void k_pool(const float* __restrict__ h, const int* __restrict__ batch,
                       const float* __restrict__ orb, float* __restrict__ cnt,
                       float* __restrict__ sumb, unsigned* __restrict__ maxb,
                       float* __restrict__ ot, int total) {
  int t = blockIdx.x * blockDim.x + threadIdx.x;
  if (t >= total) return;
  int n = t >> 8, c = t & 255;
  int bg = batch[n];
  float v = h[t];
  atomicAdd(&sumb[(size_t)bg * CH + c], v);
  atomicMax(&maxb[(size_t)bg * CH + c], fkey(v));
  if (c == 0) atomicAdd(&cnt[bg], 1.f);
  if (c < 4) atomicAdd(&ot[bg * 4 + c], orb[(size_t)n * 4 + c]);
}

__global__ void k_finalize_pool(const float* __restrict__ sumb, const unsigned* __restrict__ maxb,
                                const float* __restrict__ cnt, float* __restrict__ comb, int total) {
  int t = blockIdx.x * blockDim.x + threadIdx.x;
  if (t >= total) return;
  int g = t >> 8, c = t & 255;
  float s = sumb[t];
  comb[(size_t)g * 1024 + c]        = s / cnt[g];       // mean
  comb[(size_t)g * 1024 + CH + c]   = funkey(maxb[t]);  // max
  comb[(size_t)g * 1024 + 2*CH + c] = s;                // sum
}

// ---------------------------------------------------------------------------
// Host driver
// ---------------------------------------------------------------------------
static inline void gemm(hipStream_t s, const float* A, const float* B, const float* bias,
                        float* C, int M, int N, int K, int ldc, int act) {
  dim3 g((N + BN - 1) / BN, (M + BM - 1) / BM);
  k_gemm<<<g, 256, 0, s>>>(A, B, bias, C, M, N, K, ldc, act);
}
static inline void zero(hipStream_t s, void* p, size_t n) {
  k_zero<<<(unsigned)((n + 255) / 256), 256, 0, s>>>((float*)p, n);
}
#define EW(kern, total, ...) kern<<<((total) + 255) / 256, 256, 0, stream>>>(__VA_ARGS__)

extern "C" void kernel_launch(void* const* d_in, const int* in_sizes, int n_in,
                              void* d_out, int out_size, void* d_ws, size_t ws_size,
                              hipStream_t stream) {
  (void)in_sizes; (void)n_in; (void)out_size; (void)ws_size;

  const int*   x      = (const int*)d_in[0];
  const int*   eidx   = (const int*)d_in[1];
  const float* eattr  = (const float*)d_in[2];
  const float* energ  = (const float*)d_in[3];
  const float* orb    = (const float*)d_in[4];
  const int*   batch  = (const int*)d_in[5];
  const int*   src    = eidx;
  const int*   dst    = eidx + NE;

  // params flattened in jax pytree (sorted-key) order after positional inputs
  const float* const* P = (const float* const*)(d_in + 6);
  const float* edge_emb_W = P[0];
  const float* edge_emb_b = P[1];
  const float* fc_W1 = P[2];  const float* fc_W2 = P[3];
  const float* fc_b1 = P[4];  const float* fc_b2 = P[5];
  const float* fe_W1 = P[6];  const float* fe_W2 = P[7];
  const float* fe_b1 = P[8];  const float* fe_b2 = P[9];
  const float* fo_W  = P[10]; const float* fo_b  = P[11]; const float* fo_g = P[12];
  // layers: base 13, 37 leaves each (conv 21 + vn_real 8 + vn_virtual 8)
  const float* node_emb    = P[13 + 3 * 37];
  const float* virtual_emb = P[13 + 3 * 37 + 1];

  // workspace bump allocator (~1.3 GB total)
  char* wsp = (char*)d_ws;
  auto alloc = [&](size_t nf) -> float* {
    float* p = (float*)wsp;
    wsp += ((nf * sizeof(float) + 255) & ~(size_t)255);
    return p;
  };
  float* h    = alloc((size_t)NN * CH);
  float* ht   = alloc((size_t)NN * CH);
  float* skip = alloc((size_t)NN * CH);
  float* o    = alloc((size_t)NN * CH);
  float* q    = alloc((size_t)NN * HC);
  float* kk   = alloc((size_t)NN * HC);
  float* vv   = alloc((size_t)NN * HC);
  float* eft  = alloc((size_t)NE * HC);
  float* efeat= alloc((size_t)NE * CH);
  float* rbf  = alloc((size_t)NE * EDIM);
  float* alpha= alloc((size_t)NE * NH);
  unsigned* amax = (unsigned*)alloc((size_t)NN * NH);
  float* asum = alloc((size_t)NN * NH);
  float* m_a  = alloc((size_t)NE * NH * TC);
  float* m_b  = alloc((size_t)NE * NH * TC);
  float* m_c  = alloc((size_t)NE * NH * CH);
  float* agg  = alloc((size_t)NN * HC);
  float* vx   = alloc((size_t)NG * CH);
  float* vtmp = alloc((size_t)NG * CH);
  float* cnt  = alloc(NG);
  float* sumb = alloc((size_t)NG * CH);
  unsigned* maxb = (unsigned*)alloc((size_t)NG * CH);
  float* ot   = alloc((size_t)NG * 4);
  float* comb = alloc((size_t)NG * 1024);
  float* tmp1 = alloc((size_t)NG * 256);
  float* tmpo = alloc((size_t)NG * 128);
  float* hid  = alloc((size_t)NG * 1024);

  const int EH = NE * NH;

  // ---- embeddings ----
  EW(k_embed, NN * CH, x, node_emb, h, NN * CH);
  EW(k_rbf, NE * EDIM, eattr, rbf, NE * EDIM);
  gemm(stream, rbf, edge_emb_W, edge_emb_b, efeat, NE, CH, EDIM, CH, 0);
  EW(k_vxinit, NG * CH, virtual_emb, vx, NG * CH);

  // ---- 3 Matformer layers ----
  for (int L = 0; L < 3; ++L) {
    const float* const* Cv = P + 13 + L * 37;
    const float *Wc = Cv[0], *We = Cv[1], *Wk = Cv[2], *Wml = Cv[3], *Wmu = Cv[4],
                *Wq = Cv[5], *Ws = Cv[6], *Wv = Cv[7], *b_ml = Cv[8], *bc = Cv[9],
                *bk = Cv[10], *bml = Cv[11], *bmu = Cv[12], *bn_b = Cv[13],
                *bn_g = Cv[14], *bn_m = Cv[15], *bn_v = Cv[16], *bq = Cv[17],
                *bs = Cv[18], *bv = Cv[19], *g_ml = Cv[20];
    const float* const* Vr = Cv + 21;  // vn_real:    W1 W2 b1 b2 bb1 bb2 g1 g2
    const float* const* Vw = Cv + 29;  // vn_virtual

    zero(stream, amax, (size_t)NN * NH);
    zero(stream, asum, (size_t)NN * NH);
    zero(stream, agg,  (size_t)NN * HC);

    gemm(stream, h, Wq, bq, q,  NN, HC, CH, HC, 0);
    gemm(stream, h, Wk, bk, kk, NN, HC, CH, HC, 0);
    gemm(stream, h, Wv, bv, vv, NN, HC, CH, HC, 0);
    gemm(stream, efeat, We, nullptr, eft, NE, HC, CH, HC, 0);

    k_alpha<<<(EH * 32 + 255) / 256, 256, 0, stream>>>(q, kk, eft, src, dst, alpha, amax, EH);
    EW(k_expsum, EH, alpha, dst, amax, asum, EH);
    EW(k_anorm, EH, alpha, dst, asum, EH);

    EW(k_build_msg, NE * NH * CH, vv, eft, alpha, src, dst, m_a, NE * NH * CH);
    gemm(stream, m_a, Wmu, bmu, m_b, EH, TC, TC, TC, 0);
    gemm(stream, m_b, Wml, bml, m_c, EH, CH, TC, CH, 0);
    k_ln<<<EH, 256, 0, stream>>>(m_c, m_c, g_ml, b_ml, CH, CH, CH, 0);
    EW(k_scatter, NE * NH * CH, m_c, dst, agg, NE * NH * CH);

    gemm(stream, agg, Wc, bc, o, NN, CH, HC, CH, 0);
    gemm(stream, h, Ws, bs, skip, NN, CH, CH, CH, 0);
    EW(k_bnsilu, NN * CH, o, skip, h, bn_g, bn_b, bn_m, bn_v, NN * CH);

    // virtual node: vx += segsum(h); vx = MLP(vx)
    EW(k_vxacc, NN * CH, h, batch, vx, NN * CH);
    gemm(stream, vx, Vw[0], Vw[2], vtmp, NG, CH, CH, CH, 0);
    k_ln<<<NG, 256, 0, stream>>>(vtmp, vtmp, Vw[6], Vw[4], CH, CH, CH, 1);
    gemm(stream, vtmp, Vw[1], Vw[3], vx, NG, CH, CH, CH, 0);
    k_ln<<<NG, 256, 0, stream>>>(vx, vx, Vw[7], Vw[5], CH, CH, CH, 1);

    // real nodes: h = MLP(h + vx[batch])
    EW(k_addvx, NN * CH, h, vx, batch, ht, NN * CH);
    gemm(stream, ht, Vr[0], Vr[2], o, NN, CH, CH, CH, 0);
    k_ln<<<NN, 256, 0, stream>>>(o, o, Vr[6], Vr[4], CH, CH, CH, 1);
    gemm(stream, o, Vr[1], Vr[3], h, NN, CH, CH, CH, 0);
    k_ln<<<NN, 256, 0, stream>>>(h, h, Vr[7], Vr[5], CH, CH, CH, 1);
  }

  // ---- readout ----
  zero(stream, cnt, NG);
  zero(stream, sumb, (size_t)NG * CH);
  zero(stream, maxb, (size_t)NG * CH);
  zero(stream, ot,   (size_t)NG * 4);
  EW(k_pool, NN * CH, h, batch, orb, cnt, sumb, maxb, ot, NN * CH);
  EW(k_finalize_pool, NG * CH, sumb, maxb, cnt, comb, NG * CH);   // comb[:, 0:768]

  // energy branch -> comb[:, 768:896]
  gemm(stream, energ, fe_W1, fe_b1, tmp1, NG, 256, 201, 256, 1);
  gemm(stream, tmp1, fe_W2, fe_b2, comb + TC, NG, 128, 256, 1024, 0);
  // orbital branch -> comb[:, 896:1024]
  gemm(stream, ot, fo_W, nullptr, tmpo, NG, 128, 4, 128, 0);
  k_ln<<<NG, 256, 0, stream>>>(tmpo, comb + TC + 128, fo_g, fo_b, 128, 128, 1024, 0);

  // final MLP -> d_out [64, 804]
  gemm(stream, comb, fc_W1, fc_b1, hid, NG, 1024, 1024, 1024, 1);
  gemm(stream, hid, fc_W2, fc_b2, (float*)d_out, NG, 804, 1024, 804, 0);
}
